// GATv2NetWeights_80479097193055
// MI455X (gfx1250) — matrix operand order repaired and checked by reading, not compiled, not run on hardware
//
#include <hip/hip_runtime.h>

typedef __attribute__((ext_vector_type(16))) _Float16 v16h;
typedef __attribute__((ext_vector_type(8)))  _Float16 v8h;
typedef __attribute__((ext_vector_type(8)))  float    v8f;

#define NN 50000           // nodes
#define NE 800000          // edges (without self loops)
#define EA (NE + NN)       // edges incl. self loops
#define DF 128             // feature dim (= H*C)
#define NL 3               // layers
#define NEG_SLOPE 0.2f
#define MEAN_BLOCKS ((NE + 1023) / 1024)

static_assert(EA % 16 == 0, "edge grid must tile exactly (16 edges per 256-thr block)");
static_assert(NN % 16 == 0, "node rows must tile exactly for WMMA");

// ---- monotone float<->uint mapping for atomic segment-max ----
static __device__ __forceinline__ unsigned fmap(float f) {
  unsigned b = __float_as_uint(f);
  return (b & 0x80000000u) ? ~b : (b | 0x80000000u);
}
static __device__ __forceinline__ float funmap(unsigned u) {
  unsigned b = (u & 0x80000000u) ? (u ^ 0x80000000u) : ~u;
  return __uint_as_float(b);
}
// native CDNA5 f32 atomic add (no return -> STOREcnt path, RMW at device scope / L2)
static __device__ __forceinline__ void atomAddF(float* a, float v) {
  asm volatile("global_atomic_add_f32 %0, %1, off scope:SCOPE_DEV"
               :: "v"(a), "v"(v) : "memory");
}

// ---- fp32 -> fp16 copy of node features ----
__global__ void kconv_x(const float* __restrict__ x, _Float16* __restrict__ xh) {
  int i = blockIdx.x * 256 + threadIdx.x;
  if (i < NN * DF) xh[i] = (_Float16)x[i];
}

// ---- weights: fp32 [L][K][N] -> fp16 transposed [mat][N][K]; mat = 2l (Wl), 2l+1 (Wr) ----
__global__ void kconv_w(const float* __restrict__ Wl, const float* __restrict__ Wr,
                        _Float16* __restrict__ WT) {
  int i = blockIdx.x * 256 + threadIdx.x;
  if (i >= 6 * DF * DF) return;
  int m = i >> 14, r = i & 16383, k = r >> 7, n = r & 127, l = m >> 1;
  float v = (m & 1) ? Wr[l * DF * DF + k * DF + n] : Wl[l * DF * DF + k * DF + n];
  WT[(size_t)m * DF * DF + n * DF + k] = (_Float16)v;
}

// ---- deterministic mean of edge_weight (two-stage tree reduction) ----
__global__ void kmean1(const float* __restrict__ ew, float* __restrict__ part) {
  __shared__ float sh[256];
  int base = blockIdx.x * 1024 + threadIdx.x;
  float s = 0.f;
#pragma unroll
  for (int j = 0; j < 4; ++j) { int i = base + j * 256; if (i < NE) s += ew[i]; }
  sh[threadIdx.x] = s; __syncthreads();
  for (int off = 128; off > 0; off >>= 1) {
    if (threadIdx.x < off) sh[threadIdx.x] += sh[threadIdx.x + off];
    __syncthreads();
  }
  if (threadIdx.x == 0) part[blockIdx.x] = sh[0];
}
__global__ void kmean2(const float* __restrict__ part, float* __restrict__ meanOut, int np) {
  __shared__ float sh[256];
  float s = 0.f;
  for (int i = threadIdx.x; i < np; i += 256) s += part[i];
  sh[threadIdx.x] = s; __syncthreads();
  for (int off = 128; off > 0; off >>= 1) {
    if (threadIdx.x < off) sh[threadIdx.x] += sh[threadIdx.x + off];
    __syncthreads();
  }
  if (threadIdx.x == 0) meanOut[0] = sh[0] / (float)NE;
}

// ---- WMMA GEMM: XL = Xh*Wl + bl, XR = Xh*Wr + br; one wave = one 16x16 tile of both ----
__global__ void kgemm(const _Float16* __restrict__ Xh,
                      const _Float16* __restrict__ WTl, const _Float16* __restrict__ WTr,
                      const float* __restrict__ bl, const float* __restrict__ br,
                      float* __restrict__ XL, float* __restrict__ XR) {
  const int lane = threadIdx.x & 31;
  const int wave = threadIdx.x >> 5;        // column tile 0..7
  const int rowT = blockIdx.x;              // row tile 0..3124 (3125*16 == 50000)
  const int m16  = lane & 15;
  const int hi   = lane >> 4;
  const int row  = rowT * 16 + m16;         // A: lane holds row m16
  const int ncol = wave * 16 + m16;         // B: lane holds column m16
  const _Float16* xrow = Xh  + (size_t)row  * DF;
  const _Float16* lrow = WTl + (size_t)ncol * DF;
  const _Float16* rrow = WTr + (size_t)ncol * DF;
  v8f accL = {}; v8f accR = {};
#pragma unroll
  for (int kc = 0; kc < 4; ++kc) {
    const int kbA = kc * 32 + hi * 8;       // A layout: K = hi*8+j | 16+hi*8+j
    const int kbB = kc * 32 + hi * 8;       // B layout: K = hi*8+j | 16+hi*8+j
    v8h a0 = *(const v8h*)(xrow + kbA);
    v8h a1 = *(const v8h*)(xrow + kbA + 16);
    v8h l0 = *(const v8h*)(lrow + kbB);
    v8h l1 = *(const v8h*)(lrow + kbB + 16);
    v8h r0 = *(const v8h*)(rrow + kbB);
    v8h r1 = *(const v8h*)(rrow + kbB + 16);
    v16h a, bL, bR;
#pragma unroll
    for (int j = 0; j < 8; ++j) {
      a[j] = a0[j];  a[j + 8]  = a1[j];
      bL[j] = l0[j]; bL[j + 8] = l1[j];
      bR[j] = r0[j]; bR[j + 8] = r1[j];
    }
    accL = __builtin_amdgcn_wmma_f32_16x16x32_f16(false, a, false, bL, (short)0, accL, false, false);
    accR = __builtin_amdgcn_wmma_f32_16x16x32_f16(false, a, false, bR, (short)0, accR, false, false);
  }
  const float bli = bl[ncol], bri = br[ncol];
#pragma unroll
  for (int i = 0; i < 8; ++i) {             // D: lane -> N=m16, VGPR i -> M = hi*8+i
    size_t off = ((size_t)(rowT * 16 + hi * 8 + i)) * DF + ncol;
    XL[off] = accL[i] + bli;
    XR[off] = accR[i] + bri;
  }
}

// ---- zero accumulators per layer ----
__global__ void kzero(float* __restrict__ outv, unsigned* __restrict__ smax,
                      float* __restrict__ den) {
  int i = blockIdx.x * 256 + threadIdx.x;
  if (i < NN * DF) outv[i] = 0.f;
  if (i < NN * 4) { smax[i] = 0u; den[i] = 0.f; }
}

// ---- pass 1: wave per 2 edges -> per-head scores + atomic segment max ----
__global__ void kedge_score(const float* __restrict__ XL, const float* __restrict__ XR,
                            const int* __restrict__ srcv, const int* __restrict__ dstv,
                            const float* __restrict__ ew, const float* __restrict__ meanv,
                            const float* __restrict__ We, const float* __restrict__ att,
                            float* __restrict__ score, unsigned* __restrict__ smax) {
  const int lane = threadIdx.x & 31;
  const int wid = blockIdx.x * 8 + (threadIdx.x >> 5);
  int s[2], d[2]; float w[2];
#pragma unroll
  for (int j = 0; j < 2; ++j) {
    int e = wid * 2 + j;                    // EA % 16 == 0 -> always in range
    if (e < NE) { s[j] = srcv[e]; d[j] = dstv[e]; w[j] = ew[e]; }
    else        { s[j] = d[j] = e - NE; w[j] = meanv[0]; }
  }
  // prefetch second edge's rows (one lane per 16B -> covers the full 512B row)
  __builtin_prefetch(XL + (size_t)s[1] * DF + lane * 4, 0, 3);
  __builtin_prefetch(XR + (size_t)d[1] * DF + lane * 4, 0, 3);
#pragma unroll
  for (int j = 0; j < 2; ++j) {
    const int e = wid * 2 + j;
    const float* xls = XL + (size_t)s[j] * DF;
    const float* xrd = XR + (size_t)d[j] * DF;
    float sc[4];
#pragma unroll
    for (int k = 0; k < 4; ++k) {           // head k, channel = lane
      int idx = k * 32 + lane;
      float m = xls[idx] + xrd[idx] + w[j] * We[idx];
      m = (m > 0.f) ? m : NEG_SLOPE * m;    // leaky_relu
      sc[k] = m * att[idx];
    }
#pragma unroll
    for (int off = 16; off > 0; off >>= 1)
#pragma unroll
      for (int k = 0; k < 4; ++k) sc[k] += __shfl_xor(sc[k], off, 32);
    if (lane < 4) {
      score[e * 4 + lane] = sc[lane];
      atomicMax(&smax[(size_t)d[j] * 4 + lane], fmap(sc[lane]));
    }
  }
}

// ---- pass 2: exp(score - max), accumulate denominator ----
__global__ void kedge_soft(float* __restrict__ score, const int* __restrict__ dstv,
                           const unsigned* __restrict__ smax, float* __restrict__ den) {
  int idx = blockIdx.x * 256 + threadIdx.x;
  if (idx >= EA * 4) return;
  int e = idx >> 2, h = idx & 3;
  int d = (e < NE) ? dstv[e] : (e - NE);
  float ex = expf(score[idx] - funmap(smax[(size_t)d * 4 + h]));
  score[idx] = ex;
  atomAddF(&den[(size_t)d * 4 + h], ex);
}

// ---- pass 3: wave per 2 edges -> alpha * xl[src] scattered into out[dst] ----
__global__ void kedge_scatter(const float* __restrict__ XL, const float* __restrict__ ex,
                              const int* __restrict__ srcv, const int* __restrict__ dstv,
                              const float* __restrict__ den, float* __restrict__ outv) {
  const int lane = threadIdx.x & 31;
  const int wid = blockIdx.x * 8 + (threadIdx.x >> 5);
  int s[2], d[2];
#pragma unroll
  for (int j = 0; j < 2; ++j) {
    int e = wid * 2 + j;
    if (e < NE) { s[j] = srcv[e]; d[j] = dstv[e]; }
    else        { s[j] = d[j] = e - NE; }
  }
  __builtin_prefetch(XL + (size_t)s[1] * DF + lane * 4, 0, 3);
#pragma unroll
  for (int j = 0; j < 2; ++j) {
    const int e = wid * 2 + j;
    const float* xls = XL + (size_t)s[j] * DF;
    float* od = outv + (size_t)d[j] * DF;
#pragma unroll
    for (int k = 0; k < 4; ++k) {
      float alpha = ex[e * 4 + k] / den[(size_t)d[j] * 4 + k];  // broadcast loads
      atomAddF(&od[k * 32 + lane], alpha * xls[k * 32 + lane]);
    }
  }
}

// ---- node update: relu(out + bias); emit fp32 for next stage + fp16 for next GEMM ----
__global__ void knode(const float* __restrict__ outv, const float* __restrict__ bias,
                      float* __restrict__ xf, _Float16* __restrict__ xh) {
  int i = blockIdx.x * 256 + threadIdx.x;
  if (i >= NN * DF) return;
  float v = outv[i] + bias[i & 127];
  v = (v > 0.f) ? v : 0.f;
  xf[i] = v;
  xh[i] = (_Float16)v;
}

// ---- final projection: y = x @ Wf + bf (OUT = 1) ----
__global__ void kfinal(const float* __restrict__ xf, const float* __restrict__ Wf,
                       const float* __restrict__ bf, float* __restrict__ y) {
  __shared__ float wf[DF];
  if (threadIdx.x < DF) wf[threadIdx.x] = Wf[threadIdx.x];
  __syncthreads();
  int n = blockIdx.x * 256 + threadIdx.x;
  if (n >= NN) return;
  const float* xr = xf + (size_t)n * DF;
  float s = 0.f;
#pragma unroll 8
  for (int i = 0; i < DF; ++i) s += xr[i] * wf[i];
  y[n] = s + bf[0];
}

extern "C" void kernel_launch(void* const* d_in, const int* in_sizes, int n_in,
                              void* d_out, int out_size, void* d_ws, size_t ws_size,
                              hipStream_t stream) {
  (void)in_sizes; (void)n_in; (void)out_size; (void)ws_size;
  const float* x    = (const float*)d_in[0];
  const int*   ei   = (const int*)  d_in[1];
  const float* ew   = (const float*)d_in[2];
  const float* Wl   = (const float*)d_in[3];
  const float* bl   = (const float*)d_in[4];
  const float* Wr   = (const float*)d_in[5];
  const float* br   = (const float*)d_in[6];
  const float* We   = (const float*)d_in[7];
  const float* att  = (const float*)d_in[8];
  const float* bias = (const float*)d_in[9];
  const float* Wf   = (const float*)d_in[10];
  const float* bf   = (const float*)d_in[11];
  float* y = (float*)d_out;

  char* p = (char*)d_ws;
  auto take = [&](size_t n) { char* r = p; p += (n + 255) & ~(size_t)255; return r; };
  _Float16* Xh   = (_Float16*)take((size_t)NN * DF * 2);
  _Float16* WT   = (_Float16*)take((size_t)6 * DF * DF * 2);
  float*    XL   = (float*)take((size_t)NN * DF * 4);
  float*    XR   = (float*)take((size_t)NN * DF * 4);
  float*    OUTA = (float*)take((size_t)NN * DF * 4);
  float*    XF   = (float*)take((size_t)NN * DF * 4);
  float*    SC   = (float*)take((size_t)EA * 4 * 4);
  unsigned* SMAX = (unsigned*)take((size_t)NN * 4 * 4);
  float*    DEN  = (float*)take((size_t)NN * 4 * 4);
  float*    PART = (float*)take((size_t)MEAN_BLOCKS * 4);
  float*    MEAN = (float*)take(256);

  const int* srcv = ei;
  const int* dstv = ei + NE;

  kconv_x<<<(NN * DF + 255) / 256, 256, 0, stream>>>(x, Xh);
  kconv_w<<<(6 * DF * DF + 255) / 256, 256, 0, stream>>>(Wl, Wr, WT);
  kmean1<<<MEAN_BLOCKS, 256, 0, stream>>>(ew, PART);
  kmean2<<<1, 256, 0, stream>>>(PART, MEAN, MEAN_BLOCKS);

  for (int l = 0; l < NL; ++l) {
    kgemm<<<NN / 16, 256, 0, stream>>>(Xh, WT + (size_t)(2 * l) * DF * DF,
                                       WT + (size_t)(2 * l + 1) * DF * DF,
                                       bl + l * DF, br + l * DF, XL, XR);
    kzero<<<(NN * DF + 255) / 256, 256, 0, stream>>>(OUTA, SMAX, DEN);
    kedge_score<<<EA / 16, 256, 0, stream>>>(XL, XR, srcv, dstv, ew, MEAN,
                                             We + l * DF, att + l * DF, SC, SMAX);
    kedge_soft<<<(EA * 4 + 255) / 256, 256, 0, stream>>>(SC, dstv, SMAX, DEN);
    kedge_scatter<<<EA / 16, 256, 0, stream>>>(XL, SC, srcv, dstv, DEN, OUTA);
    knode<<<(NN * DF + 255) / 256, 256, 0, stream>>>(OUTA, bias + l * DF, XF, Xh);
  }
  kfinal<<<(NN + 255) / 256, 256, 0, stream>>>(XF, Wf, bf, y);
}
